// CRSM_84378927497574
// MI455X (gfx1250) — matrix operand order repaired and checked by reading, not compiled, not run on hardware
//
#include <hip/hip_runtime.h>

typedef __attribute__((ext_vector_type(16))) _Float16 v16h;
typedef __attribute__((ext_vector_type(8)))  _Float16 v8h;
typedef __attribute__((ext_vector_type(8)))  float    v8f;

#define N_NODES 100000
#define N_EDGES 1600000
#define F_DIM   128
#define CON     64
#define HID     256
#define OUT_DIM 128
#define IN_DIM  192   // F_DIM + CON

// workspace layout (bytes)
#define SUM_OFF   0ULL
#define SUM_BYTES ((size_t)N_NODES * CON * sizeof(float))          // 25,600,000
#define DEG_OFF   (SUM_OFF + SUM_BYTES)
#define DEG_BYTES ((size_t)N_NODES * sizeof(float))                //    400,000
#define W1F_OFF   (DEG_OFF + DEG_BYTES)                            // 26,000,000 (32B aligned)
#define W1F_HALVES (16 * 6 * 32 * 16)                              // ntiles*kchunks*lanes*16
#define W2F_OFF   (W1F_OFF + (size_t)W1F_HALVES * sizeof(_Float16))
#define W2F_HALVES (8 * 8 * 32 * 16)

// ---------------------------------------------------------------------------
// Pack W1 (192x256) and W2 (256x128) f32 row-major into this kernel's WMMA B-fragment
// order (f16): frag[(ntile*KC + kc)*32 + lane][e] = W[k][n]
//   n = ntile*16 + (lane & 15)
//   k = kc*32 + (lane>>4)*16 + e        (lanes 0-15: K 0-15, lanes 16-31: K 16-31)
// ---------------------------------------------------------------------------
__global__ void prep_weights_kernel(const float* __restrict__ W1,
                                    const float* __restrict__ W2,
                                    _Float16* __restrict__ w1f,
                                    _Float16* __restrict__ w2f) {
    int t = blockIdx.x * blockDim.x + threadIdx.x;
    if (t < W1F_HALVES) {
        int e = t & 15, lane = (t >> 4) & 31, rest = t >> 9;
        int kc = rest % 6, nt = rest / 6;
        int n = nt * 16 + (lane & 15);
        int k = kc * 32 + ((lane >> 4) << 4) + e;
        w1f[t] = (_Float16)W1[k * HID + n];
    } else if (t < W1F_HALVES + W2F_HALVES) {
        int u = t - W1F_HALVES;
        int e = u & 15, lane = (u >> 4) & 31, rest = u >> 9;
        int kc = rest & 7, nt = rest >> 3;
        int n = nt * 16 + (lane & 15);
        int k = kc * 32 + ((lane >> 4) << 4) + e;
        w2f[u] = (_Float16)W2[k * OUT_DIM + n];
    }
}

// ---------------------------------------------------------------------------
// Symmetric scatter: for each edge (i,j) add conical[j] into sum[i] and
// conical[i] into sum[j]; count degrees. One thread per (edge, float4 chunk).
// sum/deg region (~26MB) is L2-resident on MI455X (192MB L2).
// ---------------------------------------------------------------------------
__global__ void scatter_kernel(const float4* __restrict__ x4,
                               const int* __restrict__ ei,
                               float* __restrict__ sum,
                               float* __restrict__ deg) {
    long long tid = (long long)blockIdx.x * blockDim.x + threadIdx.x;
    if (tid >= (long long)N_EDGES * 16) return;
    int e = (int)(tid >> 4);
    int c = (int)(tid & 15);            // float4 chunk 0..15 of the 64 conical feats
    int i = ei[e];
    int j = ei[N_EDGES + e];
    float4 vj = x4[(long long)j * (F_DIM / 4) + c];
    float4 vi = x4[(long long)i * (F_DIM / 4) + c];
    float* si = sum + (long long)i * CON + c * 4;
    float* sj = sum + (long long)j * CON + c * 4;
    atomicAdd(si + 0, vj.x); atomicAdd(si + 1, vj.y);
    atomicAdd(si + 2, vj.z); atomicAdd(si + 3, vj.w);
    atomicAdd(sj + 0, vi.x); atomicAdd(sj + 1, vi.y);
    atomicAdd(sj + 2, vi.z); atomicAdd(sj + 3, vi.w);
    if (c == 0) {
        atomicAdd(deg + i, 1.0f);
        atomicAdd(deg + j, 1.0f);
    }
}

// ---------------------------------------------------------------------------
// Fused 2-layer MLP with WMMA. One wave32 per 16-node tile, 4 waves per block.
// combined (f16) and h (f16) staged in LDS; weights read as packed B-frags.
// A-frag per-lane layout (16x32 f16): lane&15 = M row, hi = lane>>4,
//   elems 0..7  -> K = kc*32 + hi*16 + 0..7
//   elems 8..15 -> K = kc*32 + hi*16 + 8 + 0..7
// D-tile layout: col = lane&15, row = vgpr + hi*8.
// ---------------------------------------------------------------------------
__launch_bounds__(128)
__global__ void mlp_kernel(const float* __restrict__ x,
                           const float* __restrict__ sum,
                           const float* __restrict__ deg,
                           const v16h* __restrict__ w1f,
                           const v16h* __restrict__ w2f,
                           const float* __restrict__ b1,
                           const float* __restrict__ b2,
                           float* __restrict__ out) {
    __shared__ __align__(16) _Float16 comb[4][16][IN_DIM];  // 24 KB
    __shared__ __align__(16) _Float16 hsh[4][16][HID];      // 32 KB

    const int wave = threadIdx.x >> 5;
    const int lane = threadIdx.x & 31;
    const int tileBase = (blockIdx.x * 4 + wave) * 16;

    // ---- build combined = [x | sum/deg] as f16 in LDS (zero-pad past N) ----
    for (int t = lane; t < 16 * IN_DIM; t += 32) {
        int r = t / IN_DIM, k = t % IN_DIM;
        int node = tileBase + r;
        float v = 0.0f;
        if (node < N_NODES) {
            if (k < F_DIM) {
                v = x[(long long)node * F_DIM + k];
            } else {
                float d = fmaxf(deg[node], 1.0f);
                v = sum[(long long)node * CON + (k - F_DIM)] / d;
            }
        }
        comb[wave][r][k] = (_Float16)v;
    }
    __syncthreads();

    const int col = lane & 15;   // A: M row; D: N column
    const int hi  = lane >> 4;

    // ---- layer 1: h = relu(combined @ W1 + b1) ----
    union { v16h v; v8h h[2]; } a1[6];
    for (int kc = 0; kc < 6; ++kc) {
        int kb = kc * 32 + hi * 16;
        a1[kc].h[0] = *(const v8h*)&comb[wave][col][kb];
        a1[kc].h[1] = *(const v8h*)&comb[wave][col][kb + 8];
    }
    for (int nt = 0; nt < 16; ++nt) {
        v8f acc = {};
        for (int kc = 0; kc < 6; ++kc) {
            v16h bf = w1f[(nt * 6 + kc) * 32 + lane];
            acc = __builtin_amdgcn_wmma_f32_16x16x32_f16(
                false, a1[kc].v, false, bf, (short)0, acc, false, false);
        }
        float bias = b1[nt * 16 + col];
        for (int g = 0; g < 8; ++g) {
            float v = fmaxf(acc[g] + bias, 0.0f);
            hsh[wave][g + hi * 8][nt * 16 + col] = (_Float16)v;
        }
    }
    __syncthreads();

    // ---- layer 2: out = h @ W2 + b2 ----
    union { v16h v; v8h h[2]; } a2[8];
    for (int kc = 0; kc < 8; ++kc) {
        int kb = kc * 32 + hi * 16;
        a2[kc].h[0] = *(const v8h*)&hsh[wave][col][kb];
        a2[kc].h[1] = *(const v8h*)&hsh[wave][col][kb + 8];
    }
    for (int nt = 0; nt < 8; ++nt) {
        v8f acc = {};
        for (int kc = 0; kc < 8; ++kc) {
            v16h bf = w2f[(nt * 8 + kc) * 32 + lane];
            acc = __builtin_amdgcn_wmma_f32_16x16x32_f16(
                false, a2[kc].v, false, bf, (short)0, acc, false, false);
        }
        float bias = b2[nt * 16 + col];
        for (int g = 0; g < 8; ++g) {
            int node = tileBase + g + hi * 8;
            if (node < N_NODES)
                out[(long long)node * OUT_DIM + nt * 16 + col] = acc[g] + bias;
        }
    }
}

extern "C" void kernel_launch(void* const* d_in, const int* in_sizes, int n_in,
                              void* d_out, int out_size, void* d_ws, size_t ws_size,
                              hipStream_t stream) {
    const float* x  = (const float*)d_in[0];
    const int*   ei = (const int*)d_in[1];       // edge_index [2, E]
    const float* W1 = (const float*)d_in[2];
    const float* b1 = (const float*)d_in[3];
    const float* W2 = (const float*)d_in[4];
    const float* b2 = (const float*)d_in[5];
    float* out = (float*)d_out;

    char* ws = (char*)d_ws;
    float*    sum = (float*)(ws + SUM_OFF);
    float*    deg = (float*)(ws + DEG_OFF);
    _Float16* w1f = (_Float16*)(ws + W1F_OFF);
    _Float16* w2f = (_Float16*)(ws + W2F_OFF);

    // zero the accumulators (graph-capturable memset node)
    hipMemsetAsync(ws, 0, SUM_BYTES + DEG_BYTES, stream);

    int pthreads = W1F_HALVES + W2F_HALVES;
    prep_weights_kernel<<<(pthreads + 255) / 256, 256, 0, stream>>>(W1, W2, w1f, w2f);

    long long sthreads = (long long)N_EDGES * 16;
    int sblocks = (int)((sthreads + 255) / 256);
    scatter_kernel<<<sblocks, 256, 0, stream>>>((const float4*)x, ei, sum, deg);

    int mblocks = (N_NODES + 63) / 64;   // 64 nodes per block (4 waves x 16)
    mlp_kernel<<<mblocks, 128, 0, stream>>>(x, sum, deg,
                                            (const v16h*)w1f, (const v16h*)w2f,
                                            b1, b2, out);
}
